// Ultimate_STCN_GraphSNN_20590073217512
// MI455X (gfx1250) — matrix operand order repaired and checked by reading, not compiled, not run on hardware
//
#include <hip/hip_runtime.h>
#include <math.h>

typedef __attribute__((ext_vector_type(16))) __bf16 v16bf;
typedef __attribute__((ext_vector_type(8)))  __bf16 v8bf;
typedef __attribute__((ext_vector_type(8)))  float  v8f;

static constexpr int Bn = 64;
static constexpr int Tn = 1152;
static constexpr int Cn = 64;
static constexpr int Fn = 8;
static constexpr int PADl = 15;   // SAME pad-low for kernel 32
static constexpr int NCHUNK = 48; // T / 24

// ---------------------------------------------------------------------------
// k0: pack WMMA A-fragments (conv weights, M = output feature o) + M2 collapse
// A-matrix 16x32 bf16 per-lane layout packed here (contiguous k, B matches; wave32):
//   lane l<16 : M=l,    e0..15 -> K=0..15
//   lane l>=16: M=l-16, e0..15 -> K=16..31
// Rows M=8..15 (invalid o) are zero so D rows 8..15 are exact zeros.
// ---------------------------------------------------------------------------
__global__ void __launch_bounds__(256) k0_prep(
    const float* __restrict__ L,      // (B,64,64)
    const float* __restrict__ convk,  // (32,1,8,8) HWIO
    const float* __restrict__ sw,     // (64,4)
    __bf16* __restrict__ wfrag,       // (8 ksteps, 32 lanes, 16)
    float* __restrict__ M2,           // (B,64 m,4 s)
    float* __restrict__ partials)     // 128 floats
{
  const int tid = threadIdx.x;
  if (blockIdx.x == 0) {
    const int ks = tid >> 5, lane = tid & 31;
    const int o = lane & 15;
    const int kbase = (lane & 16) ? 16 : 0;
#pragma unroll
    for (int e = 0; e < 16; ++e) {
      const int kl = kbase + e; // K local in [0,32)
      const int kh = ks * 4 + (kl >> 3);                   // tap
      const int i  = kl & 7;                               // input feature
      const float v = (o < 8) ? convk[kh * 64 + i * 8 + o] : 0.f;
      wfrag[(ks * 32 + lane) * 16 + e] = (__bf16)v;
    }
    if (tid < 128) partials[tid] = 0.f;
  } else {
    const int b = blockIdx.x - 1;
    const int m = tid >> 2, s = tid & 3;
    float acc = 0.f;
    for (int c = 0; c < 64; ++c)
      acc = fmaf(sw[c * 4 + s], L[(b * 64 + c) * 64 + m], acc);
    M2[(b * 64 + m) * 4 + s] = acc;
  }
}

// ---------------------------------------------------------------------------
// k1: X (B,T,C,F) f32 -> Xbf (B,C,T,F) bf16 via LDS tile transpose
// ---------------------------------------------------------------------------
__global__ void __launch_bounds__(256) k1_convert(
    const float* __restrict__ X, __bf16* __restrict__ Xbf)
{
  __shared__ __bf16 sb[32 * 264]; // [c][t*8+i], padded stride 264
  const int blk = blockIdx.x;
  const int b  = blk / 72;
  const int r  = blk % 72;
  const int t0 = (r >> 1) * 32;
  const int c0 = (r & 1) * 32;
  const int tid = threadIdx.x;
#pragma unroll
  for (int k = 0; k < 32; ++k) {
    const int flat = tid + k * 256;       // (t, c, i)
    const int t = flat >> 8;
    const int rem = flat & 255;
    const int c = rem >> 3, i = rem & 7;
    const float v = X[((b * Tn + t0 + t) * 64 + (c0 + c)) * 8 + i];
    sb[c * 264 + t * 8 + i] = (__bf16)v;
  }
  __syncthreads();
#pragma unroll
  for (int k = 0; k < 32; ++k) {
    const int flat = tid + k * 256;       // (c, t*8+i)
    const int c = flat >> 8;
    const int rem = flat & 255;
    Xbf[((b * 64 + c0 + c) * Tn + t0) * 8 + rem] = sb[c * 264 + rem];
  }
}

// ---------------------------------------------------------------------------
// k2: conv(32 taps, 8->8) via bf16 WMMA + in-register LayerNorm + ReLU + M2
// One wave = one 16-column t-subtile; loops over all 64 channels m.
// WMMA mapping: M = o (8 of 16, rest zero), N = t columns, K = (tap,i).
// D layout => lane l<16 holds ann[t0+l, o=0..7] in cc[0..7]: LN is in-lane.
// B (X) 32x16 bf16 lane layout: lane l<16 -> col t0+l, e -> K=e (taps ks*4+{0,1})
//                               lane l>=16 -> col t0+(l-16), K=16+e (taps +{2,3})
// ---------------------------------------------------------------------------
__global__ void __launch_bounds__(256) k2_conv(
    const __bf16* __restrict__ Xbf,    // (B,64,T,8)
    const __bf16* __restrict__ wfrag,  // (8,32,16)
    const float* __restrict__ M2,      // (B,64,4)
    const float* __restrict__ lnS, const float* __restrict__ lnBv,
    float* __restrict__ snn)           // (B,T,32)
{
  const int b    = blockIdx.x / 9;
  const int tile = blockIdx.x % 9;
  const int tid  = threadIdx.x;
  const int wave = tid >> 5;
  const int lane = tid & 31;
  const int t0   = tile * 128 + wave * 16;
  const int tcol = lane & 15;
  const int tapb = (lane & 16) ? 2 : 0;  // first tap of this lane's K-run
  // load all 8 weight fragments (resident for entire kernel)
  v16bf wf[8];
#pragma unroll
  for (int ks = 0; ks < 8; ++ks) {
    const v8bf* bp = (const v8bf*)(wfrag + (ks * 32 + lane) * 16);
    const v8bf lo = bp[0], hi8 = bp[1];
    wf[ks] = __builtin_shufflevector(lo, hi8, 0,1,2,3,4,5,6,7,8,9,10,11,12,13,14,15);
  }
  float gS[8], gB[8];
#pragma unroll
  for (int f = 0; f < 8; ++f) { gS[f] = lnS[f]; gB[f] = lnBv[f]; }
  float acc[32];                        // [s*8+f] for this lane's t
#pragma unroll
  for (int u = 0; u < 32; ++u) acc[u] = 0.f;

#pragma unroll 2
  for (int m = 0; m < 64; ++m) {
    const float* wm = M2 + (b * 64 + m) * 4;
    const float w0 = wm[0], w1 = wm[1], w2 = wm[2], w3 = wm[3];
    const __bf16* pA = Xbf + (size_t)((b * 64 + m) * Tn) * 8;
    if (m + 1 < 64) { // prefetch next channel's slab into caches
      int pr = t0 - PADl; if (pr < 0) pr = 0;
      __builtin_prefetch(Xbf + (size_t)((b * 64 + m + 1) * Tn + pr) * 8, 0, 1);
    }
    v8f cc0, cc1;
#pragma unroll
    for (int e = 0; e < 8; ++e) { cc0[e] = 0.f; cc1[e] = 0.f; }
#pragma unroll
    for (int ks = 0; ks < 8; ++ks) {
      const int r0 = t0 + tcol + ks * 4 + tapb - PADl; // tap tapb row
      const int r1 = r0 + 1;                           // tap tapb+1 row
      v8bf h0, h1;
      if (r0 >= 0 && r0 < Tn) h0 = *(const v8bf*)(pA + r0 * 8);
      else {
#pragma unroll
        for (int e = 0; e < 8; ++e) h0[e] = (__bf16)0.f;
      }
      if (r1 >= 0 && r1 < Tn) h1 = *(const v8bf*)(pA + r1 * 8);
      else {
#pragma unroll
        for (int e = 0; e < 8; ++e) h1[e] = (__bf16)0.f;
      }
      const v16bf bx = __builtin_shufflevector(h0, h1, 0,1,2,3,4,5,6,7,8,9,10,11,12,13,14,15);
      if (ks & 1)
        cc1 = __builtin_amdgcn_wmma_f32_16x16x32_bf16(false, wf[ks], false, bx, (short)0, cc1, false, false);
      else
        cc0 = __builtin_amdgcn_wmma_f32_16x16x32_bf16(false, wf[ks], false, bx, (short)0, cc0, false, false);
    }
    // epilogue (lanes 0..15 hold real data): in-register LN + ReLU + project
    float v[8], s = 0.f, q = 0.f;
#pragma unroll
    for (int f = 0; f < 8; ++f) {
      v[f] = cc0[f] + cc1[f];
      s += v[f];
      q = fmaf(v[f], v[f], q);
    }
    const float mu  = s * 0.125f;
    const float var = fmaxf(q * 0.125f - mu * mu, 0.f);
    const float inv = rsqrtf(var + 1e-6f);
#pragma unroll
    for (int f = 0; f < 8; ++f) {
      float a = (v[f] - mu) * inv * gS[f] + gB[f];
      a = fmaxf(a, 0.f);
      acc[0 * 8 + f] = fmaf(w0, a, acc[0 * 8 + f]);
      acc[1 * 8 + f] = fmaf(w1, a, acc[1 * 8 + f]);
      acc[2 * 8 + f] = fmaf(w2, a, acc[2 * 8 + f]);
      acc[3 * 8 + f] = fmaf(w3, a, acc[3 * 8 + f]);
    }
  }
  if (lane < 16) { // this lane owns snn row t = t0 + lane (32 contiguous floats)
    float* op = snn + ((size_t)b * Tn + (t0 + tcol)) * 32;
#pragma unroll
    for (int u = 0; u < 8; ++u) {
      *(float4*)(op + u * 4) =
          make_float4(acc[u * 4 + 0], acc[u * 4 + 1], acc[u * 4 + 2], acc[u * 4 + 3]);
    }
  }
}

// ---------------------------------------------------------------------------
// k3: two-layer LIF scan over T, one wave per batch element.
// Register double-buffered chunks of 24 timesteps (== L_DP).
// ---------------------------------------------------------------------------
#define LIF_CHUNK(RG, TC)                                                      \
  {                                                                            \
    _Pragma("unroll") for (int t = 0; t < 24; ++t) {                           \
      const float x = RG[t];                                                   \
      mem1 = mem1 * 0.8f + x;                                                  \
      const float sp1 = (mem1 - 0.5f > 0.f) ? 1.f : 0.f;                       \
      mem1 = mem1 - sp1 * 0.5f;  s1 += sp1;                                    \
      mem2 = mem2 * 0.9f + sp1;                                                \
      const float sp2 = (mem2 - 0.5f > 0.f) ? 1.f : 0.f;                       \
      mem2 = mem2 - sp2 * 0.5f;  s2 += sp2;                                    \
      memo = memo * 0.95f + sp2;                                               \
      pot += memo;                                                             \
      if (t == 11) { flat[fb + (TC) * 64 + j] = pot * (1.f / 12.f); pot = 0.f; } \
      if (t == 23) { flat[fb + (TC) * 64 + 32 + j] = pot * (1.f / 12.f); pot = 0.f; } \
    }                                                                          \
  }

__global__ void __launch_bounds__(32) k3_scan(
    const float* __restrict__ snn, float* __restrict__ flat,
    float* __restrict__ partials)
{
  const int b = blockIdx.x;
  const int j = threadIdx.x;
  const int fb = b * 3072;
  const float* src = snn + (size_t)b * Tn * 32 + j;
  float mem1 = 0.f, mem2 = 0.f, memo = 0.f;
  float s1 = 0.f, s2 = 0.f, pot = 0.f;
  float rgA[24], rgB[24];
#pragma unroll
  for (int t = 0; t < 24; ++t) rgA[t] = src[t * 32];
#pragma unroll 1
  for (int tc = 0; tc < NCHUNK; tc += 2) {
    const float* p1 = src + (size_t)(tc + 1) * 24 * 32;
#pragma unroll
    for (int t = 0; t < 24; ++t) rgB[t] = p1[t * 32];
    LIF_CHUNK(rgA, tc);
    if (tc + 2 < NCHUNK) {
      const float* p2 = src + (size_t)(tc + 2) * 24 * 32;
#pragma unroll
      for (int t = 0; t < 24; ++t) rgA[t] = p2[t * 32];
    }
    LIF_CHUNK(rgB, tc + 1);
  }
#pragma unroll
  for (int msk = 1; msk < 32; msk <<= 1) {
    s1 += __shfl_xor(s1, msk, 32);
    s2 += __shfl_xor(s2, msk, 32);
  }
  if (j == 0) { partials[b * 2] = s1; partials[b * 2 + 1] = s2; }
}

// ---------------------------------------------------------------------------
// k4: gelu head + logits + firing rate
// ---------------------------------------------------------------------------
__global__ void __launch_bounds__(32) k4_head(
    const float* __restrict__ flat,
    const float* __restrict__ W1, const float* __restrict__ b1,
    const float* __restrict__ W2, const float* __restrict__ b2,
    const float* __restrict__ partials, float* __restrict__ out)
{
  __shared__ float y[32];
  const int b = blockIdx.x;
  const int k = threadIdx.x;
  const float* fbp = flat + b * 3072;
  float acc = b1[k];
  for (int r = 0; r < 3072; ++r) acc = fmaf(fbp[r], W1[r * 32 + k], acc);
  const float x = acc;
  y[k] = 0.5f * x * (1.f + tanhf(0.7978845608028654f * (x + 0.044715f * x * x * x)));
  __syncthreads();
  if (k < 4) {
    float l = b2[k];
#pragma unroll
    for (int r = 0; r < 32; ++r) l = fmaf(y[r], W2[r * 4 + k], l);
    out[b * 4 + k] = l;
  }
  if (b == 0 && k == 0) {
    float S = 0.f;
    for (int i = 0; i < 128; ++i) S += partials[i]; // fixed order: deterministic
    out[256] = S / (2.f * (float)Bn * (float)Tn * 32.f);
  }
}

// ---------------------------------------------------------------------------
extern "C" void kernel_launch(void* const* d_in, const int* in_sizes, int n_in,
                              void* d_out, int out_size, void* d_ws, size_t ws_size,
                              hipStream_t stream)
{
  (void)in_sizes; (void)n_in; (void)out_size; (void)ws_size;
  const float* L_norm = (const float*)d_in[0];
  const float* X_seq  = (const float*)d_in[1];
  // d_in[2] = deterministic flag (unused by the math)
  const float* convk  = (const float*)d_in[3];
  const float* ln_s   = (const float*)d_in[4];
  const float* ln_b   = (const float*)d_in[5];
  const float* sw     = (const float*)d_in[6];
  const float* W1     = (const float*)d_in[7];
  const float* b1     = (const float*)d_in[8];
  const float* W2     = (const float*)d_in[9];
  const float* b2     = (const float*)d_in[10];
  float* out = (float*)d_out;

  char* ws = (char*)d_ws;
  size_t off = 0;
  auto carve = [&](size_t bytes) {
    size_t o = off;
    off = (off + bytes + 255) & ~(size_t)255;
    return o;
  };
  __bf16* Xbf     = (__bf16*)(ws + carve((size_t)Bn * Cn * Tn * Fn * 2)); // 75.5 MB
  __bf16* wfrag   = (__bf16*)(ws + carve(8 * 32 * 16 * 2));
  float*  M2      = (float*)(ws + carve((size_t)Bn * 64 * 4 * 4));
  float*  snn     = (float*)(ws + carve((size_t)Bn * Tn * 32 * 4));       // 9.4 MB
  float*  flat    = (float*)(ws + carve((size_t)Bn * 3072 * 4));
  float*  partial = (float*)(ws + carve(128 * 4));

  k0_prep<<<dim3(1 + Bn), dim3(256), 0, stream>>>(L_norm, convk, sw, wfrag, M2, partial);
  k1_convert<<<dim3(Bn * 72), dim3(256), 0, stream>>>(X_seq, Xbf);
  k2_conv<<<dim3(Bn * 9), dim3(256), 0, stream>>>(Xbf, wfrag, M2, ln_s, ln_b, snn);
  k3_scan<<<dim3(Bn), dim3(32), 0, stream>>>(snn, flat, partial);
  k4_head<<<dim3(Bn), dim3(32), 0, stream>>>(flat, W1, b1, W2, b2, partial, out);
}
